// GraphNetBlock_45019847197000
// MI455X (gfx1250) — compile-verified, matrix operand order checked by reading, not run on hardware
//
#include <hip/hip_runtime.h>
#include <hip/hip_bf16.h>

// ---------------------------------------------------------------------------
// GraphNetBlock forward for MI455X (gfx1250, wave32, WMMA bf16).
// Bandwidth-bound: all dense GEMMs use v_wmma_f32_16x16x32_bf16 with fused
// epilogues. Weights are pre-packed into WMMA B-fragment order so each lane
// fetches its 16 fragment elements with one 32-byte coalesced load (two
// global_load_b128) instead of 16 scalar u16 loads + v_mov_b16 packing.
// A row-tiles staged in LDS (ds_load_b128 fragments). Segment softmax and
// scatter-sum use f32 global atomics.
// ---------------------------------------------------------------------------

typedef __bf16 bf16_t;
typedef __attribute__((ext_vector_type(16))) __bf16 v16bf;
typedef __attribute__((ext_vector_type(2)))  __bf16 bf16x2;
typedef __attribute__((ext_vector_type(8)))  float  v8f;

#define K_LB __launch_bounds__(256)

static constexpr int Nn  = 10000;   // nodes
static constexpr int Ee  = 160000;  // edges
static constexpr int Dd  = 256;     // feature dim
static constexpr int HDm = 512;     // H*QK == H*V

// ---------------- element-wise helpers ----------------
__global__ void K_LB k_f32_to_bf16(const float* __restrict__ s,
                                   bf16_t* __restrict__ d, int n) {
  int i = blockIdx.x * 256 + threadIdx.x;
  if (i < n) d[i] = (bf16_t)s[i];
}

__global__ void K_LB k_zero_u32(unsigned* __restrict__ p, int n) {
  int i = blockIdx.x * 256 + threadIdx.x;
  if (i < n) p[i] = 0u;
}

// K index inside a bf16 A/B fragment (ISA 7.12.2, 16-bit 16x32 layout):
// elements 0..7 -> K = e + 8*halfwave ; elements 8..15 -> K = e + 8 + 8*halfwave
__device__ __forceinline__ int frag_k(int e, int hi) {
  return e + (e >= 8 ? 8 : 0) + 8 * hi;
}

// Pack f32 weight [K x NOUT] (row-major) into bf16 WMMA B-fragment order:
// P[(((nt*(K/32) + kt)*32) + lane)*16 + e] = W[kt*32 + frag_k(e,lane>>4)][nt*16 + (lane&15)]
// One thread per (tile,lane) -> writes 32 contiguous bytes.
__global__ void K_LB k_pack_b(const float* __restrict__ W, bf16_t* __restrict__ P,
                              int K, int NOUT) {
  int tid = blockIdx.x * 256 + threadIdx.x;
  int KT = K >> 5;
  int total = KT * (NOUT >> 4) * 32;
  if (tid >= total) return;
  int lane = tid & 31;
  int tile = tid >> 5;            // nt*KT + kt
  int kt = tile % KT;
  int nt = tile / KT;
  int hi = lane >> 4, nl = lane & 15;
  bf16_t* out = P + (size_t)tid * 16;
  #pragma unroll
  for (int e = 0; e < 16; ++e) {
    int k = kt * 32 + frag_k(e, hi);
    out[e] = (bf16_t)W[(size_t)k * NOUT + nt * 16 + nl];
  }
}

// ---------------- GEMM 1 of GLU FFN: [M,256] @ [256,1024] -> GLU -> bf16 [M,512]
// Wp: packed fragments, K=256 (KT=8), NOUT=1024 (NT=64)
__global__ void K_LB k_gemm_glu(const bf16_t* __restrict__ A,
                                const bf16_t* __restrict__ Wp,
                                const float*  __restrict__ bias,  // [1024]
                                bf16_t* __restrict__ T,           // [M,512]
                                int M) {
  __shared__ bf16_t As[16 * 256];
  const int row0 = blockIdx.x * 16;
  const int wave = threadIdx.x >> 5;
  const int lane = threadIdx.x & 31;
  for (int i = threadIdx.x; i < 512; i += 256) {
    int bi = i * 8;
    int r = bi >> 8, c = bi & 255;
    int gr = row0 + r;
    uint4 v{};
    if (gr < M) v = *reinterpret_cast<const uint4*>(&A[gr * 256 + c]);
    *reinterpret_cast<uint4*>(&As[bi]) = v;
  }
  __syncthreads();
  const int hi = lane >> 4, nl = lane & 15;
  const int n0 = blockIdx.y * 128 + wave * 16;     // 0..511 (the "a" half)
  const int nt_a = n0 >> 4;
  const int nt_g = (n0 + 512) >> 4;
  const v16bf* WpV = reinterpret_cast<const v16bf*>(Wp);
  v8f acc_a = {}; v8f acc_g = {};
  #pragma unroll
  for (int kt = 0; kt < 8; ++kt) {
    v16bf af;
    #pragma unroll
    for (int e = 0; e < 16; ++e)
      af[e] = As[nl * 256 + kt * 32 + frag_k(e, hi)];
    v16bf ba = WpV[(nt_a * 8 + kt) * 32 + lane];
    v16bf bg = WpV[(nt_g * 8 + kt) * 32 + lane];
    acc_a = __builtin_amdgcn_wmma_f32_16x16x32_bf16(false, af, false, ba,
                                                    (short)0, acc_a, false, false);
    acc_g = __builtin_amdgcn_wmma_f32_16x16x32_bf16(false, af, false, bg,
                                                    (short)0, acc_g, false, false);
  }
  const int nc = n0 + nl;
  const float bba = bias[nc];
  const float bbg = bias[nc + 512];
  #pragma unroll
  for (int r = 0; r < 8; ++r) {
    int gr = row0 + r + 8 * hi;
    if (gr < M) {
      float av = acc_a[r] + bba;
      float gv = acc_g[r] + bbg;
      T[gr * 512 + nc] = (bf16_t)(av * (gv > 0.f ? gv : 0.f));
    }
  }
}

// ---------------- projection: [M,256] @ [256,512] + b -> bf16 [M,512]
// Wp: packed, K=256 (KT=8), NOUT=512 (NT=32)
__global__ void K_LB k_gemm_proj(const bf16_t* __restrict__ A,
                                 const bf16_t* __restrict__ Wp,
                                 const float*  __restrict__ bias, // [512]
                                 bf16_t* __restrict__ Out,        // [M,512]
                                 int M) {
  __shared__ bf16_t As[16 * 256];
  const int row0 = blockIdx.x * 16;
  const int wave = threadIdx.x >> 5;
  const int lane = threadIdx.x & 31;
  for (int i = threadIdx.x; i < 512; i += 256) {
    int bi = i * 8;
    int r = bi >> 8, c = bi & 255;
    int gr = row0 + r;
    uint4 v{};
    if (gr < M) v = *reinterpret_cast<const uint4*>(&A[gr * 256 + c]);
    *reinterpret_cast<uint4*>(&As[bi]) = v;
  }
  __syncthreads();
  const int hi = lane >> 4, nl = lane & 15;
  const int n0 = blockIdx.y * 128 + wave * 16;
  const int nt = n0 >> 4;
  const v16bf* WpV = reinterpret_cast<const v16bf*>(Wp);
  v8f acc = {};
  #pragma unroll
  for (int kt = 0; kt < 8; ++kt) {
    v16bf af;
    #pragma unroll
    for (int e = 0; e < 16; ++e)
      af[e] = As[nl * 256 + kt * 32 + frag_k(e, hi)];
    v16bf bf = WpV[(nt * 8 + kt) * 32 + lane];
    acc = __builtin_amdgcn_wmma_f32_16x16x32_bf16(false, af, false, bf,
                                                  (short)0, acc, false, false);
  }
  const int nc = n0 + nl;
  const float bb = bias[nc];
  #pragma unroll
  for (int r = 0; r < 8; ++r) {
    int gr = row0 + r + 8 * hi;
    if (gr < M) Out[gr * 512 + nc] = (bf16_t)(acc[r] + bb);
  }
}

// ---------------- [M,512] @ [512,256] + b, optional relu, ReZero residual
// Wp: packed, K=512 (KT=16), NOUT=256 (NT=16)
__global__ void K_LB k_gemm_k512_resid(const bf16_t* __restrict__ A,
                                       const bf16_t* __restrict__ Wp,
                                       const float*  __restrict__ bias, // [256]
                                       const float*  __restrict__ base, // [M,256]
                                       float*        __restrict__ outf, // [M,256]
                                       bf16_t*       __restrict__ outb, // nullable
                                       const float*  __restrict__ alpha,
                                       int M, int relu) {
  __shared__ bf16_t As[16 * 512];
  const int row0 = blockIdx.x * 16;
  const int wave = threadIdx.x >> 5;
  const int lane = threadIdx.x & 31;
  for (int i = threadIdx.x; i < 1024; i += 256) {
    int bi = i * 8;
    int r = bi >> 9, c = bi & 511;
    int gr = row0 + r;
    uint4 v{};
    if (gr < M) v = *reinterpret_cast<const uint4*>(&A[gr * 512 + c]);
    *reinterpret_cast<uint4*>(&As[bi]) = v;
  }
  __syncthreads();
  const int hi = lane >> 4, nl = lane & 15;
  const int n0 = blockIdx.y * 128 + wave * 16;   // grid.y == 2 -> 256 cols
  const int nt = n0 >> 4;
  const v16bf* WpV = reinterpret_cast<const v16bf*>(Wp);
  v8f acc = {};
  #pragma unroll
  for (int kt = 0; kt < 16; ++kt) {
    v16bf af;
    #pragma unroll
    for (int e = 0; e < 16; ++e)
      af[e] = As[nl * 512 + kt * 32 + frag_k(e, hi)];
    v16bf bf = WpV[(nt * 16 + kt) * 32 + lane];
    acc = __builtin_amdgcn_wmma_f32_16x16x32_bf16(false, af, false, bf,
                                                  (short)0, acc, false, false);
  }
  const int nc = n0 + nl;
  const float bb = bias[nc];
  const float al = alpha[0];
  #pragma unroll
  for (int r = 0; r < 8; ++r) {
    int gr = row0 + r + 8 * hi;
    if (gr < M) {
      float o = acc[r] + bb;
      if (relu) o = o > 0.f ? o : 0.f;
      o = base[gr * 256 + nc] + al * o;
      outf[gr * 256 + nc] = o;
      if (outb) outb[gr * 256 + nc] = (bf16_t)o;
    }
  }
}

// ---------------- attention scores: per (edge, head) dot products ----------
__device__ __forceinline__ void unpack8(uint4 w, float* f) {
  unsigned u[4] = {w.x, w.y, w.z, w.w};
  #pragma unroll
  for (int t = 0; t < 4; ++t) {
    bf16x2 p = __builtin_bit_cast(bf16x2, u[t]);
    f[2 * t]     = (float)p[0];
    f[2 * t + 1] = (float)p[1];
  }
}

__global__ void K_LB k_scores(const bf16_t* __restrict__ q,
                              const bf16_t* __restrict__ kk,
                              const bf16_t* __restrict__ ek,
                              const int* __restrict__ src,
                              const int* __restrict__ dst,
                              float* __restrict__ s_nn,
                              float* __restrict__ s_ne, int E) {
  int idx = blockIdx.x * 256 + threadIdx.x;
  if (idx >= E * 8) return;
  int e = idx >> 3, h = idx & 7;
  int s = src[e], d = dst[e];
  const uint4* qr = reinterpret_cast<const uint4*>(q  + (size_t)d * 512 + h * 64);
  const uint4* kr = reinterpret_cast<const uint4*>(kk + (size_t)s * 512 + h * 64);
  const uint4* er = reinterpret_cast<const uint4*>(ek + (size_t)e * 512 + h * 64);
  float nn = 0.f, ne = 0.f;
  #pragma unroll
  for (int t = 0; t < 8; ++t) {
    float qf[8], kf[8], ef[8];
    unpack8(qr[t], qf);
    unpack8(kr[t], kf);
    unpack8(er[t], ef);
    #pragma unroll
    for (int j = 0; j < 8; ++j) {
      nn += qf[j] * kf[j];
      ne += qf[j] * ef[j];
    }
  }
  s_nn[idx] = nn;
  s_ne[idx] = ne;
}

// monotone float<->uint map for atomic segment-max
__device__ __forceinline__ unsigned fmap(float f) {
  unsigned u = __float_as_uint(f);
  return (u & 0x80000000u) ? ~u : (u | 0x80000000u);
}
__device__ __forceinline__ float funmap(unsigned u) {
  return __uint_as_float((u & 0x80000000u) ? (u & 0x7fffffffu) : ~u);
}

__global__ void K_LB k_seg_max(const float* __restrict__ s,
                               const int* __restrict__ dst,
                               unsigned* __restrict__ m, int E) {
  int idx = blockIdx.x * 256 + threadIdx.x;
  if (idx >= E * 8) return;
  int e = idx >> 3, h = idx & 7;
  atomicMax(&m[dst[e] * 8 + h], fmap(s[idx]));
}

__global__ void K_LB k_exp_den(float* __restrict__ s,
                               const int* __restrict__ dst,
                               const unsigned* __restrict__ m,
                               float* __restrict__ den, int E) {
  int idx = blockIdx.x * 256 + threadIdx.x;
  if (idx >= E * 8) return;
  int e = idx >> 3, h = idx & 7;
  int d = dst[e];
  float ex = __expf(s[idx] - funmap(m[d * 8 + h]));
  s[idx] = ex;
  atomicAdd(&den[d * 8 + h], ex);
}

// wv = (a_n2n + a_n2e)*v[src] + a_n2e*ev ; scatter-sum into z[dst]
__global__ void K_LB k_wv_scatter(const float* __restrict__ s_nn,
                                  const float* __restrict__ s_ne,
                                  const float* __restrict__ den_nn,
                                  const float* __restrict__ den_ne,
                                  const int* __restrict__ src,
                                  const int* __restrict__ dst,
                                  const bf16_t* __restrict__ v,
                                  const bf16_t* __restrict__ ev,
                                  float* __restrict__ z, int E) {
  int idx = blockIdx.x * 256 + threadIdx.x;
  if (idx >= E * 512) return;
  int e = idx >> 9, c = idx & 511, h = c >> 6;
  int s = src[e], d = dst[e];
  float a_nn = s_nn[e * 8 + h] / den_nn[d * 8 + h];
  float a_ne = s_ne[e * 8 + h] / den_ne[d * 8 + h];
  float val = (a_nn + a_ne) * (float)v[(size_t)s * 512 + c]
            + a_ne * (float)ev[(size_t)e * 512 + c];
  atomicAdd(&z[(size_t)d * 512 + c], val);
}

// per-column sum / sumsq over z (N x 512): 16384 threads = 512 cols x 32 groups
__global__ void K_LB k_col_stats(const float* __restrict__ z,
                                 float* __restrict__ colsum,
                                 float* __restrict__ colsq, int N) {
  int tid = blockIdx.x * 256 + threadIdx.x;   // 64 blocks x 256
  int c = tid & 511;
  int rg = tid >> 9;                          // 0..31
  float s = 0.f, sq = 0.f;
  for (int r = rg; r < N; r += 32) {
    float v = z[(size_t)r * 512 + c];
    s += v; sq += v * v;
  }
  atomicAdd(&colsum[c], s);
  atomicAdd(&colsq[c], sq);
}

__global__ void K_LB k_finalize_stats(const float* __restrict__ colsum,
                                      const float* __restrict__ colsq,
                                      float* __restrict__ mu,
                                      float* __restrict__ rstd, int N) {
  int c = blockIdx.x * 256 + threadIdx.x;
  if (c >= 512) return;
  float m = colsum[c] / (float)N;
  float var = colsq[c] / (float)N - m * m;
  mu[c] = m;
  rstd[c] = rsqrtf(var + 1e-5f);
}

__global__ void K_LB k_zn(const float* __restrict__ z,
                          const float* __restrict__ mu,
                          const float* __restrict__ rstd,
                          const float* __restrict__ gamma,
                          const float* __restrict__ beta,
                          bf16_t* __restrict__ zn, int total) {
  int i = blockIdx.x * 256 + threadIdx.x;
  if (i >= total) return;
  int c = i & 511;
  zn[i] = (bf16_t)((z[i] - mu[c]) * rstd[c] * gamma[c] + beta[c]);
}

// ---------------------------------------------------------------------------
extern "C" void kernel_launch(void* const* d_in, const int* in_sizes, int n_in,
                              void* d_out, int out_size, void* d_ws, size_t ws_size,
                              hipStream_t stream) {
  (void)in_sizes; (void)n_in; (void)out_size; (void)ws_size;

  const float* n_in_f = (const float*)d_in[0];
  const float* e_in_f = (const float*)d_in[1];
  const int*   src    = (const int*)d_in[2];
  const int*   dst    = (const int*)d_in[3];
  const float* nf1_w1 = (const float*)d_in[4];  const float* nf1_b1 = (const float*)d_in[5];
  const float* nf1_w2 = (const float*)d_in[6];  const float* nf1_b2 = (const float*)d_in[7];
  const float* ef1_w1 = (const float*)d_in[8];  const float* ef1_b1 = (const float*)d_in[9];
  const float* ef1_w2 = (const float*)d_in[10]; const float* ef1_b2 = (const float*)d_in[11];
  const float* nf2_w1 = (const float*)d_in[12]; const float* nf2_b1 = (const float*)d_in[13];
  const float* nf2_w2 = (const float*)d_in[14]; const float* nf2_b2 = (const float*)d_in[15];
  const float* ef2_w1 = (const float*)d_in[16]; const float* ef2_b1 = (const float*)d_in[17];
  const float* ef2_w2 = (const float*)d_in[18]; const float* ef2_b2 = (const float*)d_in[19];
  const float* w_q  = (const float*)d_in[20];   const float* b_q  = (const float*)d_in[21];
  const float* w_k  = (const float*)d_in[22];   const float* b_k  = (const float*)d_in[23];
  const float* w_v  = (const float*)d_in[24];   const float* b_v  = (const float*)d_in[25];
  /* w_eq/b_eq (26,27) are dead code in the reference */
  const float* w_ek = (const float*)d_in[28];   const float* b_ek = (const float*)d_in[29];
  const float* w_ev = (const float*)d_in[30];   const float* b_ev = (const float*)d_in[31];
  const float* gn_gamma = (const float*)d_in[32];
  const float* gn_beta  = (const float*)d_in[33];
  const float* mix_w = (const float*)d_in[34];  const float* mix_b = (const float*)d_in[35];
  const float* alpha_n = (const float*)d_in[36];
  const float* alpha_e = (const float*)d_in[37];

  float* out_n = (float*)d_out;
  float* out_e = out_n + (size_t)Nn * Dd;

  // -------- workspace layout --------
  char* ws = (char*)d_ws;
  size_t off = 0;
  auto alloc = [&](size_t bytes) -> char* {
    char* p = ws + off;
    off = (off + bytes + 255) & ~(size_t)255;
    return p;
  };
  bf16_t* nf1w1b = (bf16_t*)alloc((size_t)256 * 1024 * 2);
  bf16_t* nf1w2b = (bf16_t*)alloc((size_t)512 * 256 * 2);
  bf16_t* ef1w1b = (bf16_t*)alloc((size_t)256 * 1024 * 2);
  bf16_t* ef1w2b = (bf16_t*)alloc((size_t)512 * 256 * 2);
  bf16_t* nf2w1b = (bf16_t*)alloc((size_t)256 * 1024 * 2);
  bf16_t* nf2w2b = (bf16_t*)alloc((size_t)512 * 256 * 2);
  bf16_t* ef2w1b = (bf16_t*)alloc((size_t)256 * 1024 * 2);
  bf16_t* ef2w2b = (bf16_t*)alloc((size_t)512 * 256 * 2);
  bf16_t* wqb  = (bf16_t*)alloc((size_t)256 * 512 * 2);
  bf16_t* wkb  = (bf16_t*)alloc((size_t)256 * 512 * 2);
  bf16_t* wvb  = (bf16_t*)alloc((size_t)256 * 512 * 2);
  bf16_t* wekb = (bf16_t*)alloc((size_t)256 * 512 * 2);
  bf16_t* wevb = (bf16_t*)alloc((size_t)256 * 512 * 2);
  bf16_t* mixwb = (bf16_t*)alloc((size_t)512 * 256 * 2);

  bf16_t* n_bf = (bf16_t*)alloc((size_t)Nn * Dd * 2);     // n / n1 / n2 (bf16)
  bf16_t* e_bf = (bf16_t*)alloc((size_t)Ee * Dd * 2);     // e / e1 (bf16)
  bf16_t* big1 = (bf16_t*)alloc((size_t)Ee * HDm * 2);    // t_edge1, then ek
  bf16_t* big2 = (bf16_t*)alloc((size_t)Ee * HDm * 2);    // ev, then t_edge2
  bf16_t* q_bf = (bf16_t*)alloc((size_t)Nn * HDm * 2);    // t_node1 / q / t_node2
  bf16_t* k_bf = (bf16_t*)alloc((size_t)Nn * HDm * 2);    // k, then zn
  bf16_t* v_bf = (bf16_t*)alloc((size_t)Nn * HDm * 2);    // v
  float* s_nn = (float*)alloc((size_t)Ee * 8 * 4);
  float* s_ne = (float*)alloc((size_t)Ee * 8 * 4);
  unsigned* m_nn = (unsigned*)alloc((size_t)Nn * 8 * 4);
  unsigned* m_ne = (unsigned*)alloc((size_t)Nn * 8 * 4);
  float* den_nn = (float*)alloc((size_t)Nn * 8 * 4);
  float* den_ne = (float*)alloc((size_t)Nn * 8 * 4);
  float* z = (float*)alloc((size_t)Nn * HDm * 4);
  float* stats = (float*)alloc((size_t)4 * 512 * 4);
  float* colsum = stats, *colsq = stats + 512, *mu = stats + 1024, *rstd = stats + 1536;

  auto cdiv = [](int a, int b) { return (a + b - 1) / b; };
  // pack weight [K x NOUT] -> fragment order ((K*NOUT)/16 threads)
  #define PACK(s_, d_, K_, N_) \
    k_pack_b<<<cdiv((K_) * (N_) / 16, 256), 256, 0, stream>>>((s_), (d_), (K_), (N_))
  #define CONV(s_, d_, n_) \
    k_f32_to_bf16<<<cdiv((n_), 256), 256, 0, stream>>>((s_), (d_), (n_))

  // -------- weight packing + feature conversion to bf16 --------
  PACK(nf1_w1, nf1w1b, 256, 1024);  PACK(nf1_w2, nf1w2b, 512, 256);
  PACK(ef1_w1, ef1w1b, 256, 1024);  PACK(ef1_w2, ef1w2b, 512, 256);
  PACK(nf2_w1, nf2w1b, 256, 1024);  PACK(nf2_w2, nf2w2b, 512, 256);
  PACK(ef2_w1, ef2w1b, 256, 1024);  PACK(ef2_w2, ef2w2b, 512, 256);
  PACK(w_q, wqb, 256, 512);  PACK(w_k, wkb, 256, 512);  PACK(w_v, wvb, 256, 512);
  PACK(w_ek, wekb, 256, 512);  PACK(w_ev, wevb, 256, 512);
  PACK(mix_w, mixwb, 512, 256);
  CONV(n_in_f, n_bf, Nn * Dd);
  CONV(e_in_f, e_bf, Ee * Dd);

  const dim3 blk(256);
  const dim3 gN16(cdiv(Nn, 16), 4), gN16x2(cdiv(Nn, 16), 2);
  const dim3 gE16(cdiv(Ee, 16), 4), gE16x2(cdiv(Ee, 16), 2);

  // -------- node FFN1: n1 = n + a_n * pwff(n) --------
  k_gemm_glu<<<gN16, blk, 0, stream>>>(n_bf, nf1w1b, nf1_b1, q_bf, Nn);
  k_gemm_k512_resid<<<gN16x2, blk, 0, stream>>>(q_bf, nf1w2b, nf1_b2, n_in_f,
                                                out_n, n_bf, alpha_n, Nn, 0);
  // -------- edge FFN1: e1 = e + a_e * pwff(e) --------
  k_gemm_glu<<<gE16, blk, 0, stream>>>(e_bf, ef1w1b, ef1_b1, big1, Ee);
  k_gemm_k512_resid<<<gE16x2, blk, 0, stream>>>(big1, ef1w2b, ef1_b2, e_in_f,
                                                out_e, e_bf, alpha_e, Ee, 0);

  // -------- projections (bf16 outputs) --------
  k_gemm_proj<<<gN16, blk, 0, stream>>>(n_bf, wqb, b_q, q_bf, Nn);
  k_gemm_proj<<<gN16, blk, 0, stream>>>(n_bf, wkb, b_k, k_bf, Nn);
  k_gemm_proj<<<gN16, blk, 0, stream>>>(n_bf, wvb, b_v, v_bf, Nn);
  k_gemm_proj<<<gE16, blk, 0, stream>>>(e_bf, wekb, b_ek, big1, Ee);
  k_gemm_proj<<<gE16, blk, 0, stream>>>(e_bf, wevb, b_ev, big2, Ee);

  // -------- attention scores --------
  k_scores<<<cdiv(Ee * 8, 256), blk, 0, stream>>>(q_bf, k_bf, big1, src, dst,
                                                  s_nn, s_ne, Ee);

  // -------- zero accumulators --------
  k_zero_u32<<<cdiv(Nn * 8, 256), blk, 0, stream>>>(m_nn, Nn * 8);
  k_zero_u32<<<cdiv(Nn * 8, 256), blk, 0, stream>>>(m_ne, Nn * 8);
  k_zero_u32<<<cdiv(Nn * 8, 256), blk, 0, stream>>>((unsigned*)den_nn, Nn * 8);
  k_zero_u32<<<cdiv(Nn * 8, 256), blk, 0, stream>>>((unsigned*)den_ne, Nn * 8);
  k_zero_u32<<<cdiv(Nn * HDm, 256), blk, 0, stream>>>((unsigned*)z, Nn * HDm);
  k_zero_u32<<<cdiv(1024, 256), blk, 0, stream>>>((unsigned*)stats, 1024);

  // -------- segment softmax (max -> exp/sum; normalization fused in scatter)
  k_seg_max<<<cdiv(Ee * 8, 256), blk, 0, stream>>>(s_nn, dst, m_nn, Ee);
  k_seg_max<<<cdiv(Ee * 8, 256), blk, 0, stream>>>(s_ne, dst, m_ne, Ee);
  k_exp_den<<<cdiv(Ee * 8, 256), blk, 0, stream>>>(s_nn, dst, m_nn, den_nn, Ee);
  k_exp_den<<<cdiv(Ee * 8, 256), blk, 0, stream>>>(s_ne, dst, m_ne, den_ne, Ee);

  // -------- weighted values, scatter-sum into z --------
  k_wv_scatter<<<cdiv(Ee * 512, 256), blk, 0, stream>>>(
      s_nn, s_ne, den_nn, den_ne, src, dst, v_bf, big2, z, Ee);

  // -------- graph-wise feature norm + mix proj + relu + residual --------
  k_col_stats<<<64, blk, 0, stream>>>(z, colsum, colsq, Nn);
  k_finalize_stats<<<2, blk, 0, stream>>>(colsum, colsq, mu, rstd, Nn);
  k_zn<<<cdiv(Nn * HDm, 256), blk, 0, stream>>>(z, mu, rstd, gn_gamma, gn_beta,
                                                k_bf, Nn * HDm);
  k_gemm_k512_resid<<<gN16x2, blk, 0, stream>>>(k_bf, mixwb, mix_b, out_n,
                                                out_n, n_bf, alpha_n, Nn, 1);

  // -------- node FFN2 --------
  k_gemm_glu<<<gN16, blk, 0, stream>>>(n_bf, nf2w1b, nf2_b1, q_bf, Nn);
  k_gemm_k512_resid<<<gN16x2, blk, 0, stream>>>(q_bf, nf2w2b, nf2_b2, out_n,
                                                out_n, nullptr, alpha_n, Nn, 0);
  // -------- edge FFN2 --------
  k_gemm_glu<<<gE16, blk, 0, stream>>>(e_bf, ef2w1b, ef2_b1, big2, Ee);
  k_gemm_k512_resid<<<gE16x2, blk, 0, stream>>>(big2, ef2w2b, ef2_b2, out_e,
                                                out_e, nullptr, alpha_e, Ee, 0);
}
